// PDN_43628277793362
// MI455X (gfx1250) — matrix operand order repaired and checked by reading, not compiled, not run on hardware
//
#include <hip/hip_runtime.h>

typedef __attribute__((ext_vector_type(16))) _Float16 v16h;
typedef __attribute__((ext_vector_type(8)))  float    v8f;

#define N_NODES 100000
#define N_EDGES 3200000
#define IN_FEAT 128
#define HID 64
#define EDGE_FEAT 16
#define PDN_HID 32

// xor-swizzle across lanes: group-of-32 mode, and_mask=0x1f, or=0, xor=m
#define SXOR(v, imm) __int_as_float(__builtin_amdgcn_ds_swizzle(__float_as_int(v), imm))
// butterfly merge: lanes with (lane&M)==0 keep the A-family, others the B-family
#define BMERGE(R, A, B, M, IMM)                    \
  {                                                \
    float c_ = (lane & (M)) ? (B) : (A);           \
    float d_ = (lane & (M)) ? (A) : (B);           \
    R = c_ + SXOR(d_, IMM);                        \
  }

// ---------------------------------------------------------------------------
// Edge-weight MLP: sigmoid(relu(edge_x @ We1 + be1) @ We2 + be2) -> ew[E]
// plus atomic accumulation of ew into deg[dst]. One wave = 16 edges,
// two v_wmma_f32_16x16x32_f16 (K zero-padded 16->32).
// ---------------------------------------------------------------------------
__global__ __launch_bounds__(256) void edge_mlp_wmma(
    const float* __restrict__ EX, const float* __restrict__ We1,
    const float* __restrict__ be1, const float* __restrict__ We2,
    const float* __restrict__ be2, const int* __restrict__ dst,
    float* __restrict__ ew, float* __restrict__ deg) {
  const int lane = threadIdx.x & 31;
  const int wave = threadIdx.x >> 5;
  const int n  = lane & 15;
  const int hi = lane >> 4;
  const int e0 = (blockIdx.x * 8 + wave) * 16;

  // A: row M = n (edge e0+n); halves 0..7 = K hi*8.., halves 8..15 = K16+ pad
  v16h a = {};
  {
    const float* ex = EX + (size_t)(e0 + n) * EDGE_FEAT + hi * 8;
#pragma unroll
    for (int i = 0; i < 8; ++i) a[i] = (_Float16)ex[i];
  }
  // B: col N = n; halves 0..7 = K hi*8.., halves 8..15 = K16+ pad
  v16h b0 = {}, b1 = {};
  {
#pragma unroll
    for (int i = 0; i < 8; ++i) {
      b0[i] = (_Float16)We1[(hi * 8 + i) * PDN_HID + n];
      b1[i] = (_Float16)We1[(hi * 8 + i) * PDN_HID + 16 + n];
    }
  }
  v8f c0 = {}, c1 = {};
  c0 = __builtin_amdgcn_wmma_f32_16x16x32_f16(false, a, false, b0, (short)0, c0, false, false);
  c1 = __builtin_amdgcn_wmma_f32_16x16x32_f16(false, a, false, b1, (short)0, c1, false, false);

  // c0[v][lane] = hid[M = v + 8*hi, N = n]; c1 covers hidden 16..31.
  const float b1a = be1[n], b1b = be1[16 + n];
  const float w2a = We2[n], w2b = We2[16 + n];
  const float bb2 = be2[0];

#define TCOMP(v) (fmaxf(c0[v] + b1a, 0.f) * w2a + fmaxf(c1[v] + b1b, 0.f) * w2b)
  float t0 = TCOMP(0), t1 = TCOMP(1), t2 = TCOMP(2), t3 = TCOMP(3);
  float t4 = TCOMP(4), t5 = TCOMP(5), t6 = TCOMP(6), t7 = TCOMP(7);
#undef TCOMP
  // Butterfly merge-reduce 8 values over the 16 lanes of each half: 8 swizzles
  float u0, u1, u2, u3, w0, w1, r;
  BMERGE(u0, t0, t1, 1, 0x041f)
  BMERGE(u1, t2, t3, 1, 0x041f)
  BMERGE(u2, t4, t5, 1, 0x041f)
  BMERGE(u3, t6, t7, 1, 0x041f)
  BMERGE(w0, u0, u1, 2, 0x081f)
  BMERGE(w1, u2, u3, 2, 0x081f)
  BMERGE(r,  w0, w1, 4, 0x101f)
  r += SXOR(r, 0x201f);
  // lane l now holds the sum for edge (l&7) + 8*hi (duplicated at l^8)
  if ((lane & 8) == 0) {
    int e = e0 + (lane & 7) + 8 * hi;
    float s = 1.0f / (1.0f + __expf(-(r + bb2)));
    ew[e] = s;
    atomicAdd(&deg[dst[e]], s);
  }
}

// ---------------------------------------------------------------------------
// Node GEMM: C[N,64] = A[N,K] @ W[K,64], f32->f16 on the fly, K fully
// unrolled (templated), optional ReLU fused into the A operand.
// ---------------------------------------------------------------------------
template <int K, int RELU>
__global__ __launch_bounds__(256) void gemm_wmma(
    const float* __restrict__ A, const float* __restrict__ W,
    float* __restrict__ C) {
  const int lane = threadIdx.x & 31;
  const int wave = threadIdx.x >> 5;
  const int n  = lane & 15;
  const int hi = lane >> 4;
  const int r0 = (blockIdx.x * 2 + (wave >> 2)) * 16;
  const int c0 = (wave & 3) * 16;

  v8f acc = {};
  const float* arow = A + (size_t)(r0 + n) * K;
#pragma unroll
  for (int kb = 0; kb < K; kb += 32) {
    v16h a, b;
#pragma unroll
    for (int i = 0; i < 8; ++i) {
      float v0 = arow[kb + hi * 8 + i];
      float v1 = arow[kb + 16 + hi * 8 + i];
      if (RELU) { v0 = fmaxf(v0, 0.f); v1 = fmaxf(v1, 0.f); }
      a[i]     = (_Float16)v0;
      a[8 + i] = (_Float16)v1;
    }
#pragma unroll
    for (int i = 0; i < 16; ++i)
      b[i] = (_Float16)W[(size_t)(kb + hi * 8 + i + (i & 8)) * HID + c0 + n];
    acc = __builtin_amdgcn_wmma_f32_16x16x32_f16(false, a, false, b, (short)0, acc, false, false);
  }
  float* crow = C + (size_t)(r0 + 8 * hi) * HID + c0 + n;
#pragma unroll
  for (int v = 0; v < 8; ++v) crow[(size_t)v * HID] = acc[v];
}

// ---------------------------------------------------------------------------
// CSR-by-destination build (count -> single-block scan -> place)
// ---------------------------------------------------------------------------
__global__ void zero_f(float* __restrict__ p, int n) {
  int i = blockIdx.x * blockDim.x + threadIdx.x;
  if (i < n) p[i] = 0.0f;
}
__global__ void zero_i(int* __restrict__ p, int n) {
  int i = blockIdx.x * blockDim.x + threadIdx.x;
  if (i < n) p[i] = 0;
}
__global__ void count_k(const int* __restrict__ dst, int* __restrict__ cnt, int e_count) {
  int e = blockIdx.x * blockDim.x + threadIdx.x;
  if (e < e_count) atomicAdd(&cnt[dst[e]], 1);
}
// single-workgroup inclusive->exclusive scan over N counts (tiny: 100k ints)
__global__ __launch_bounds__(1024) void scan_k(const int* __restrict__ cnt,
                                               int* __restrict__ rowptr,
                                               int* __restrict__ cur, int n) {
  __shared__ int sdata[1024];
  __shared__ int carry_s;
  if (threadIdx.x == 0) carry_s = 0;
  __syncthreads();
  for (int base = 0; base < n; base += 1024) {
    int i = base + threadIdx.x;
    int v = (i < n) ? cnt[i] : 0;
    sdata[threadIdx.x] = v;
    __syncthreads();
#pragma unroll
    for (int off = 1; off < 1024; off <<= 1) {
      int t = (threadIdx.x >= off) ? sdata[threadIdx.x - off] : 0;
      __syncthreads();
      sdata[threadIdx.x] += t;
      __syncthreads();
    }
    int excl = sdata[threadIdx.x] - v + carry_s;
    if (i < n) { rowptr[i] = excl; cur[i] = excl; }
    __syncthreads();
    if (threadIdx.x == 1023) carry_s += sdata[1023];
    __syncthreads();
  }
  if (threadIdx.x == 0) rowptr[n] = carry_s;
}
__global__ void place_k(const int* __restrict__ src, const int* __restrict__ dst,
                        const float* __restrict__ norm, int* __restrict__ cur,
                        int* __restrict__ csrc, float* __restrict__ cnorm, int e_count) {
  int e = blockIdx.x * blockDim.x + threadIdx.x;
  if (e < e_count) {
    int p = atomicAdd(&cur[dst[e]], 1);
    csrc[p] = src[e];
    cnorm[p] = norm[e];
  }
}

__global__ void dinv_k(const float* __restrict__ deg, float* __restrict__ dinv, int n) {
  int i = blockIdx.x * blockDim.x + threadIdx.x;
  if (i < n) dinv[i] = rsqrtf(deg[i] + 1.0f);
}
__global__ void norm_k(const int* __restrict__ src, const int* __restrict__ dst,
                       const float* __restrict__ ew, const float* __restrict__ dinv,
                       float* __restrict__ norm, int e_count) {
  int e = blockIdx.x * blockDim.x + threadIdx.x;
  if (e < e_count) norm[e] = dinv[src[e]] * ew[e] * dinv[dst[e]];
}

// ---------------------------------------------------------------------------
// Atomic-free aggregation by gather: one wave per node, 2 feats/lane.
// out[n,:] = sum_p h[csrc[p],:]*cnorm[p] + h[n,:]*dinv[n]^2 + b[:]
// ---------------------------------------------------------------------------
__global__ __launch_bounds__(256) void gather_k(
    const float* __restrict__ h, const int* __restrict__ rowptr,
    const int* __restrict__ csrc, const float* __restrict__ cnorm,
    const float* __restrict__ dinv, const float* __restrict__ b,
    float* __restrict__ outp) {
  const int lane = threadIdx.x & 31;
  const int node = blockIdx.x * 8 + (threadIdx.x >> 5);
  const int beg = rowptr[node], end = rowptr[node + 1];
  float acc0 = 0.f, acc1 = 0.f;
  for (int p = beg; p < end; ++p) {
    if (p + 4 < end) {  // cover random L2 latency: global_prefetch_b8
      __builtin_prefetch(h + (size_t)csrc[p + 4] * HID + lane, 0, 1);
    }
    int s = csrc[p];
    float nr = cnorm[p];
    acc0 += h[(size_t)s * HID + lane] * nr;
    acc1 += h[(size_t)s * HID + 32 + lane] * nr;
  }
  float d = dinv[node], sn = d * d;
  acc0 += h[(size_t)node * HID + lane] * sn + b[lane];
  acc1 += h[(size_t)node * HID + 32 + lane] * sn + b[32 + lane];
  outp[(size_t)node * HID + lane] = acc0;
  outp[(size_t)node * HID + 32 + lane] = acc1;
}

// ---------------------------------------------------------------------------
extern "C" void kernel_launch(void* const* d_in, const int* in_sizes, int n_in,
                              void* d_out, int out_size, void* d_ws, size_t ws_size,
                              hipStream_t stream) {
  const float* x      = (const float*)d_in[0];
  const int*   eidx   = (const int*)d_in[1];
  const float* edge_x = (const float*)d_in[2];
  const float* We1    = (const float*)d_in[3];
  const float* be1    = (const float*)d_in[4];
  const float* We2    = (const float*)d_in[5];
  const float* be2    = (const float*)d_in[6];
  const float* Wg1    = (const float*)d_in[7];
  const float* bg1    = (const float*)d_in[8];
  const float* Wg2    = (const float*)d_in[9];
  const float* bg2    = (const float*)d_in[10];
  float* out = (float*)d_out;

  const int N = N_NODES, E = N_EDGES;
  const int* src = eidx;
  const int* dst = eidx + E;

  // workspace carve-up (4-byte words)
  float* ws     = (float*)d_ws;
  float* ew     = ws;                          // E
  float* norm   = ew + (size_t)E;              // E
  float* cnorm  = norm + (size_t)E;            // E (CSR-ordered norms)
  int*   csrc   = (int*)(cnorm + (size_t)E);   // E (CSR-ordered sources)
  float* deg    = (float*)(csrc + (size_t)E);  // N
  float* dinv   = deg + N;                     // N
  int*   cnt    = (int*)(dinv + N);            // N
  int*   rowptr = cnt + N;                     // N+1
  int*   cur    = rowptr + (N + 1);            // N
  float* h1     = (float*)(cur + N);           // N*64 (reused as z2)
  float* agg1   = h1 + (size_t)N * HID;        // N*64

  // 1) edge weights + weighted degree; CSR counting in parallel
  zero_f<<<(N + 255) / 256, 256, 0, stream>>>(deg, N);
  zero_i<<<(N + 255) / 256, 256, 0, stream>>>(cnt, N);
  edge_mlp_wmma<<<E / 128, 256, 0, stream>>>(edge_x, We1, be1, We2, be2, dst, ew, deg);
  count_k<<<E / 256, 256, 0, stream>>>(dst, cnt, E);
  dinv_k<<<(N + 255) / 256, 256, 0, stream>>>(deg, dinv, N);
  norm_k<<<E / 256, 256, 0, stream>>>(src, dst, ew, dinv, norm, E);
  scan_k<<<1, 1024, 0, stream>>>(cnt, rowptr, cur, N);
  place_k<<<E / 256, 256, 0, stream>>>(src, dst, norm, cur, csrc, cnorm, E);

  // 2) layer 1: h1 = x @ Wg1 ; agg1 = gather + self-loop + bias (atomic-free)
  gemm_wmma<IN_FEAT, 0><<<N / 32, 256, 0, stream>>>(x, Wg1, h1);
  gather_k<<<N / 8, 256, 0, stream>>>(h1, rowptr, csrc, cnorm, dinv, bg1, agg1);

  // 3) layer 2: z = relu(agg1) @ Wg2 (ReLU fused); out = gather(z)
  gemm_wmma<HID, 1><<<N / 32, 256, 0, stream>>>(agg1, Wg2, h1 /*z2*/);
  gather_k<<<N / 8, 256, 0, stream>>>(h1, rowptr, csrc, cnorm, dinv, bg2, out);
}
